// CrossMamba_53171695125430
// MI455X (gfx1250) — matrix operand order repaired and checked by reading, not compiled, not run on hardware
//
#include <hip/hip_runtime.h>

typedef __bf16 bf16;
typedef bf16  v16bf __attribute__((ext_vector_type(16)));
typedef bf16  v8bf  __attribute__((ext_vector_type(8)));
typedef float v8f   __attribute__((ext_vector_type(8)));

// ---- model dims ----
#define BN   2
#define TT   4
#define LSEQ 448          // visible tokens per batch
#define MR   (BN*LSEQ)    // 896 GEMM rows for layer stage
#define DM   768
#define DI   1536
#define NS   16
#define RK   48
#define RKP  64           // padded dt K
#define PWD  80           // RK + 2*NS
#define KC   4
#define DEPTH 16
#define NV   196
#define NA   24
#define NTOT 896

__device__ __forceinline__ bf16 f2b(float x) {
  unsigned u = __builtin_bit_cast(unsigned, x);
  unsigned r = u + 0x7FFFu + ((u >> 16) & 1u);     // round-to-nearest-even
  return __builtin_bit_cast(bf16, (unsigned short)(r >> 16));
}

__device__ __forceinline__ v16bf ldfrag_a(const bf16* p, int half) {
  // A lane layout: lanes 0-15 K[0..7]&K[16..23], lanes 16-31 K[8..15]&K[24..31]
  v8bf a0 = *(const v8bf*)(p + 8 * half);
  v8bf a1 = *(const v8bf*)(p + 16 + 8 * half);
  return __builtin_shufflevector(a0, a1, 0,1,2,3,4,5,6,7,8,9,10,11,12,13,14,15);
}
__device__ __forceinline__ v16bf ldfrag_b(const bf16* p, int half) {
  // B lane layout: lanes 0-15 K[0..7]&K[16..23], lanes 16-31 K[8..15]&K[24..31]
  v8bf b0 = *(const v8bf*)(p + 8 * half);
  v8bf b1 = *(const v8bf*)(p + 16 + 8 * half);
  return __builtin_shufflevector(b0, b1, 0,1,2,3,4,5,6,7,8,9,10,11,12,13,14,15);
}

// ============================================================
// Generic bf16 WMMA GEMM (1 tile / wave) for odd shapes.
// C[m,n] = sum_k A[m,k]*W[n,k] (+bias[n]); K % 32 == 0.
// ============================================================
#define GEMM_WAVES 8
__global__ __launch_bounds__(GEMM_WAVES*32)
void k_gemm_bf16(const bf16* __restrict__ A, const bf16* __restrict__ W,
                 const float* __restrict__ bias, float* __restrict__ C,
                 int M, int N, int K) {
  const int lane = threadIdx.x & 31;
  const int wave = threadIdx.x >> 5;
  const int tilesN = N >> 4;
  const int tilesTot = (M >> 4) * tilesN;
  const int tile = blockIdx.x * GEMM_WAVES + wave;
  if (tile >= tilesTot) return;
  const int tm = tile / tilesN, tn = tile % tilesN;
  const int r    = lane & 15;
  const int half = lane >> 4;
  const bf16* Ap = A + (size_t)(tm * 16 + r) * K;
  const bf16* Wp = W + (size_t)(tn * 16 + r) * K;
  v8f acc = {0.f,0.f,0.f,0.f,0.f,0.f,0.f,0.f};
  for (int kk = 0; kk < K; kk += 32) {
    v16bf av = ldfrag_a(Ap + kk, half);
    v16bf bv = ldfrag_b(Wp + kk, half);
    acc = __builtin_amdgcn_wmma_f32_16x16x32_bf16(false, av, false, bv,
                                                  (short)0, acc, false, false);
  }
  const float bvl = bias ? bias[tn * 16 + r] : 0.f;
  float* Cp = C + (size_t)(tm * 16 + 8 * half) * N + tn * 16 + r;
#pragma unroll
  for (int v = 0; v < 8; v++) Cp[(size_t)v * N] = acc[v] + bvl;
}

// ============================================================
// Register-blocked bf16 WMMA GEMM: one wave -> 64x64 output
// (4x4 tiles, 16 f32 accumulators). M%64==0, N%64==0, K%32==0.
// ============================================================
__global__ __launch_bounds__(GEMM_WAVES*32)
void k_gemm_bf16_4x4(const bf16* __restrict__ A, const bf16* __restrict__ W,
                     const float* __restrict__ bias, float* __restrict__ C,
                     int M, int N, int K) {
  const int lane = threadIdx.x & 31;
  const int wave = threadIdx.x >> 5;
  const int blocksN = N >> 6;
  const int blocksTot = (M >> 6) * blocksN;
  const int blk = blockIdx.x * GEMM_WAVES + wave;
  if (blk >= blocksTot) return;
  const int bm = blk / blocksN, bn = blk % blocksN;
  const int r    = lane & 15;
  const int half = lane >> 4;
  const bf16* Ap[4];
  const bf16* Wp[4];
#pragma unroll
  for (int i = 0; i < 4; i++) {
    Ap[i] = A + (size_t)(bm * 64 + i * 16 + r) * K;
    Wp[i] = W + (size_t)(bn * 64 + i * 16 + r) * K;
  }
  v8f acc[4][4];
#pragma unroll
  for (int i = 0; i < 4; i++)
#pragma unroll
    for (int j = 0; j < 4; j++)
      acc[i][j] = (v8f){0.f,0.f,0.f,0.f,0.f,0.f,0.f,0.f};

  for (int kk = 0; kk < K; kk += 32) {
    v16bf av[4], bv[4];
#pragma unroll
    for (int i = 0; i < 4; i++) av[i] = ldfrag_a(Ap[i] + kk, half);
#pragma unroll
    for (int j = 0; j < 4; j++) bv[j] = ldfrag_b(Wp[j] + kk, half);
#pragma unroll
    for (int i = 0; i < 4; i++)
#pragma unroll
      for (int j = 0; j < 4; j++)
        acc[i][j] = __builtin_amdgcn_wmma_f32_16x16x32_bf16(
            false, av[i], false, bv[j], (short)0, acc[i][j], false, false);
  }
#pragma unroll
  for (int j = 0; j < 4; j++) {
    const int n0 = bn * 64 + j * 16 + r;
    const float bvl = bias ? bias[n0] : 0.f;
#pragma unroll
    for (int i = 0; i < 4; i++) {
      float* Cp = C + (size_t)(bm * 64 + i * 16 + 8 * half) * N + n0;
#pragma unroll
      for (int v = 0; v < 8; v++) Cp[(size_t)v * N] = acc[i][j][v] + bvl;
    }
  }
}

// ============================================================
// TDM-staged GEMM: 4 waves (2x2) -> 128x128 macro-tile.
// Panels DMA'd to LDS by the Tensor Data Mover (double-buffered,
// TENSORcnt-tracked), WMMA operands read from LDS (ds_load_b128).
// LDS row stride 80B (64B data + 16B TDM pad) -> conflict-free.
// Requires M%128==0, N%128==0, K%32==0.
// ============================================================
#define PROW 40                    // LDS panel row stride in bf16 elems (80B)
#define PSZ  (128*PROW)            // one panel: 128 rows

#if defined(__has_builtin)
#if __has_builtin(__builtin_amdgcn_tensor_load_to_lds)
#define HAVE_TDM 1
#endif
#if __has_builtin(__builtin_amdgcn_s_wait_tensorcnt)
#define HAVE_TCNT 1
#endif
#endif

#ifdef HAVE_TDM
typedef unsigned u32x4 __attribute__((ext_vector_type(4)));
typedef int      i32x8 __attribute__((ext_vector_type(8)));
typedef int      i32x4 __attribute__((ext_vector_type(4)));

// DMA a 128-row x 32-col bf16 tile (row stride K elems) into LDS at
// lds_byte_off with 16B padding after each 64B row (ISA D# layout, 8.3-8.5).
// 6-arg builtin variant: (g0, g1, g2, g3, g4, cpol).
__device__ __forceinline__ void tdm_load_panel(const bf16* gsrc, unsigned lds_byte_off,
                                               int K, int rows) {
  unsigned long long ga = (unsigned long long)(uintptr_t)gsrc;
  u32x4 g0;
  g0[0] = 1u;                                            // count=1, user D#
  g0[1] = lds_byte_off;                                  // lds_addr
  g0[2] = (unsigned)(ga & 0xFFFFFFFFu);                  // global_addr[31:0]
  g0[3] = (unsigned)((ga >> 32) & 0x01FFFFFFu) | (2u << 30);  // addr[56:32] | type=2
  i32x8 g1;
  g1[0] = (1 << 16) | (1 << 20) | (3 << 22) | (3 << 25); // 2B elems, pad 4dw per 16dw
  g1[1] = (int)(((unsigned)K & 0xFFFFu) << 16);          // tensor_dim0 lo16 @bit48
  g1[2] = (int)((((unsigned)K >> 16) & 0xFFFFu) |
                (((unsigned)rows & 0xFFFFu) << 16));     // dim0 hi16 | dim1 lo16
  g1[3] = (int)((((unsigned)rows >> 16) & 0xFFFFu) |
                (32u << 16));                            // dim1 hi16 | tile_dim0=32
  g1[4] = 128;                                           // tile_dim1=128, tile_dim2=0
  g1[5] = K;                                             // tensor_dim0_stride lo32
  g1[6] = 0;
  g1[7] = 0;
  i32x4 z4 = {0, 0, 0, 0};
  i32x8 z8 = {0, 0, 0, 0, 0, 0, 0, 0};
  __builtin_amdgcn_tensor_load_to_lds(g0, g1, z4, z4, z8, 0);
}
#endif

__device__ __forceinline__ void tdm_wait0() {
#ifdef HAVE_TCNT
  __builtin_amdgcn_s_wait_tensorcnt(0);
#else
  asm volatile("s_wait_tensorcnt 0" ::: "memory");
#endif
}

__device__ __forceinline__ void mac_panels(const bf16* pA, const bf16* pW,
                                           int wm, int wn, int r, int half,
                                           v8f acc[4][4]) {
  v16bf av[4], bv[4];
#pragma unroll
  for (int i = 0; i < 4; i++) {
    const bf16* rp = pA + (wm * 64 + i * 16 + r) * PROW;
    v8bf a0 = *(const v8bf*)(rp + 8 * half);
    v8bf a1 = *(const v8bf*)(rp + 16 + 8 * half);
    av[i] = __builtin_shufflevector(a0, a1, 0,1,2,3,4,5,6,7,8,9,10,11,12,13,14,15);
  }
#pragma unroll
  for (int j = 0; j < 4; j++) {
    const bf16* rp = pW + (wn * 64 + j * 16 + r) * PROW;
    v8bf b0 = *(const v8bf*)(rp + 8 * half);
    v8bf b1 = *(const v8bf*)(rp + 16 + 8 * half);
    bv[j] = __builtin_shufflevector(b0, b1, 0,1,2,3,4,5,6,7,8,9,10,11,12,13,14,15);
  }
#pragma unroll
  for (int i = 0; i < 4; i++)
#pragma unroll
    for (int j = 0; j < 4; j++)
      acc[i][j] = __builtin_amdgcn_wmma_f32_16x16x32_bf16(
          false, av[i], false, bv[j], (short)0, acc[i][j], false, false);
}

__global__ __launch_bounds__(128)
void k_gemm_bf16_tdm(const bf16* __restrict__ A, const bf16* __restrict__ W,
                     const float* __restrict__ bias, float* __restrict__ C,
                     int M, int N, int K) {
  __shared__ __align__(16) bf16 sPan[4 * PSZ];   // A0,W0,A1,W1 (40KB)
  const int lane = threadIdx.x & 31;
  const int wave = threadIdx.x >> 5;
  const int blocksN = N >> 7;
  const int bm = blockIdx.x / blocksN, bn = blockIdx.x % blocksN;
  const int wm = wave >> 1, wn = wave & 1;
  const int r = lane & 15, half = lane >> 4;

  v8f acc[4][4];
#pragma unroll
  for (int i = 0; i < 4; i++)
#pragma unroll
    for (int j = 0; j < 4; j++)
      acc[i][j] = (v8f){0.f,0.f,0.f,0.f,0.f,0.f,0.f,0.f};

#ifdef HAVE_TDM
  // prologue: DMA first K-slice into buffer 0
  if (wave == 0) tdm_load_panel(A + (size_t)(bm * 128) * K,
                                (unsigned)(uintptr_t)(sPan + 0 * PSZ), K, M);
  if (wave == 1) tdm_load_panel(W + (size_t)(bn * 128) * K,
                                (unsigned)(uintptr_t)(sPan + 1 * PSZ), K, N);
  if (wave < 2) tdm_wait0();
  __syncthreads();
  int buf = 0;
  for (int kk = 0; kk < K; kk += 32) {
    const int nxt = buf ^ 1;
    const bool more = (kk + 32) < K;
    if (more) {   // prefetch next slice while computing on current
      if (wave == 0) tdm_load_panel(A + (size_t)(bm * 128) * K + kk + 32,
                                    (unsigned)(uintptr_t)(sPan + (2 * nxt) * PSZ), K, M);
      if (wave == 1) tdm_load_panel(W + (size_t)(bn * 128) * K + kk + 32,
                                    (unsigned)(uintptr_t)(sPan + (2 * nxt + 1) * PSZ), K, N);
    }
    mac_panels(sPan + (2 * buf) * PSZ, sPan + (2 * buf + 1) * PSZ, wm, wn, r, half, acc);
    if (more && wave < 2) tdm_wait0();
    __syncthreads();
    buf = nxt;
  }
#else
  // fallback: cooperative sync staging (512 x 16B chunks per panel, 128 threads)
  for (int kk = 0; kk < K; kk += 32) {
    for (int c = threadIdx.x; c < 512; c += 128) {
      int row = c >> 2, seg = c & 3;
      *(v8bf*)(sPan + row * PROW + seg * 8) =
          *(const v8bf*)(A + (size_t)(bm * 128 + row) * K + kk + seg * 8);
      *(v8bf*)(sPan + PSZ + row * PROW + seg * 8) =
          *(const v8bf*)(W + (size_t)(bn * 128 + row) * K + kk + seg * 8);
    }
    __syncthreads();
    mac_panels(sPan, sPan + PSZ, wm, wn, r, half, acc);
    __syncthreads();
  }
#endif

#pragma unroll
  for (int j = 0; j < 4; j++) {
    const int n0 = bn * 128 + wn * 64 + j * 16 + r;
    const float bvl = bias ? bias[n0] : 0.f;
#pragma unroll
    for (int i = 0; i < 4; i++) {
      float* Cp = C + (size_t)(bm * 128 + wm * 64 + i * 16 + 8 * half) * N + n0;
#pragma unroll
      for (int v = 0; v < 8; v++) Cp[(size_t)v * N] = acc[i][j][v] + bvl;
    }
  }
}

// ============================================================
// Elementwise / packing kernels
// ============================================================
__global__ void k_f32_to_bf16(const float* __restrict__ s, bf16* __restrict__ d, int n) {
  int i = blockIdx.x * 256 + threadIdx.x;
  if (i < n) d[i] = f2b(s[i]);
}

__global__ void k_copy_f32(const float* __restrict__ s, float* __restrict__ d, int n) {
  int i = blockIdx.x * 256 + threadIdx.x;
  if (i < n) d[i] = s[i];
}

// video patches: x_v[B,3,T,224,224] -> X[1568,768], col=(ph*16+pw)*3+c
__global__ void k_pack_video(const float* __restrict__ xv, bf16* __restrict__ X) {
  int i = blockIdx.x * 256 + threadIdx.x;
  const int n = BN * TT * NV * 768;
  if (i >= n) return;
  int r = i / 768, k = i % 768;
  int b = r / (TT * NV), t = (r / NV) % TT, pv = r % NV;
  int c = k % 3, pw = (k / 3) % 16, ph = (k / 3) / 16;
  int gh = (pv / 14) * 16 + ph, gw = (pv % 14) * 16 + pw;
  X[i] = f2b(xv[(((size_t)(b * 3 + c) * TT + t) * 224 + gh) * 224 + gw]);
}

// patch_v_w[D,3,16,16] -> Wv[768,768] in (ph,pw,c) flattening
__global__ void k_pack_wv(const float* __restrict__ w, bf16* __restrict__ W) {
  int i = blockIdx.x * 256 + threadIdx.x;
  const int n = 768 * 768;
  if (i >= n) return;
  int d = i / 768, k = i % 768;
  int c = k % 3, pw = (k / 3) % 16, ph = (k / 3) / 16;
  W[i] = f2b(w[((size_t)(d * 3 + c) * 16 + ph) * 16 + pw]);
}

// audio patches: x_a[B,64,384] -> X[192,256], row p=ti*4+fi, col=pt*16+pf
__global__ void k_pack_audio(const float* __restrict__ xa, bf16* __restrict__ X) {
  int i = blockIdx.x * 256 + threadIdx.x;
  const int n = BN * 96 * 256;
  if (i >= n) return;
  int r = i / 256, k = i % 256;
  int b = r / 96, p = r % 96, ti = p / 4, fi = p % 4;
  int pt = k / 16, pf = k % 16;
  X[i] = f2b(xa[((size_t)b * 64 + fi * 16 + pf) * 384 + ti * 16 + pt]);
}

// full token stream [B,896,768]
__global__ void k_assemble(const float* __restrict__ xv_emb, const float* __restrict__ xa_emb,
                           const float* __restrict__ pos_v, const float* __restrict__ pos_a,
                           const float* __restrict__ tpos_v, const float* __restrict__ tpos_a,
                           const float* __restrict__ cls_v, const float* __restrict__ cls_a,
                           const float* __restrict__ glob_v, const float* __restrict__ glob_a,
                           float* __restrict__ xfull) {
  int i = blockIdx.x * 256 + threadIdx.x;
  const int n = BN * NTOT * 768;
  if (i >= n) return;
  int d = i % 768, l = (i / 768) % NTOT, b = i / (768 * NTOT);
  int t = l / 224, ft = l % 224;
  float v;
  if (ft == 0)        v = cls_v[d];
  else if (ft <= 196) { int p = ft - 1;
    v = xv_emb[(size_t)((b * TT + t) * NV + p) * 768 + d] + pos_v[p * 768 + d] + tpos_v[t * 768 + d]; }
  else if (ft == 197) v = glob_v[d];
  else if (ft == 198) v = cls_a[d];
  else if (ft <= 222) { int p = ft - 199;
    v = xa_emb[(size_t)(b * 96 + t * NA + p) * 768 + d] + pos_a[p * 768 + d] + tpos_a[t * 768 + d]; }
  else                v = glob_a[d];
  xfull[i] = v;
}

__global__ void k_gather(const float* __restrict__ xfull, const int* __restrict__ idx,
                         float* __restrict__ h, float* __restrict__ res) {
  int i = blockIdx.x * 256 + threadIdx.x;
  const int n = BN * LSEQ * 768;
  if (i >= n) return;
  int d = i % 768, l = (i / 768) % LSEQ, b = i / (768 * LSEQ);
  h[i] = xfull[((size_t)b * NTOT + idx[b * LSEQ + l]) * 768 + d];
  res[i] = 0.f;
}

// res += h ; hn = bf16(RMSNorm(res) * nw) ; one block per token row
__global__ __launch_bounds__(256)
void k_res_rmsnorm(float* __restrict__ res, const float* __restrict__ h,
                   const float* __restrict__ nw, bf16* __restrict__ hn) {
  __shared__ float red[256];
  const int row = blockIdx.x;
  const float* hp = h + (size_t)row * 768;
  float* rp = res + (size_t)row * 768;
  float local[3], ss = 0.f;
#pragma unroll
  for (int j = 0; j < 3; j++) {
    int d = threadIdx.x + j * 256;
    float v = rp[d] + hp[d];
    local[j] = v; ss += v * v;
  }
  red[threadIdx.x] = ss; __syncthreads();
  for (int s = 128; s > 0; s >>= 1) {
    if (threadIdx.x < s) red[threadIdx.x] += red[threadIdx.x + s];
    __syncthreads();
  }
  float scale = rsqrtf(red[0] / 768.f + 1e-5f);
#pragma unroll
  for (int j = 0; j < 3; j++) {
    int d = threadIdx.x + j * 256;
    rp[d] = local[j];
    hn[(size_t)row * 768 + d] = f2b(local[j] * scale * nw[d]);
  }
}

// depthwise causal conv (KC=4) + silu over xi = xz[:, :1536]; optional time-reverse
__global__ void k_conv_silu(const float* __restrict__ xz, const float* __restrict__ cw,
                            const float* __restrict__ cb, float* __restrict__ xc,
                            bf16* __restrict__ xcb, int reverse) {
  int i = blockIdx.x * 256 + threadIdx.x;
  const int n = MR * DI;
  if (i >= n) return;
  int d = i % DI, l = (i / DI) % LSEQ, b = i / (DI * LSEQ);
  float acc = cb[d];
#pragma unroll
  for (int j = 0; j < KC; j++) {
    int lt = l - (KC - 1) + j;
    if (lt < 0) continue;
    int lsrc = reverse ? (LSEQ - 1 - lt) : lt;
    acc += cw[d * KC + j] * xz[(size_t)(b * LSEQ + lsrc) * (2 * DI) + d];
  }
  float s = acc / (1.f + __expf(-acc));
  xc[i] = s; xcb[i] = f2b(s);
}

// dtr = proj[:, :48] -> bf16 [896,64] zero-padded
__global__ void k_pad_dtr(const float* __restrict__ proj, bf16* __restrict__ d) {
  int i = blockIdx.x * 256 + threadIdx.x;
  const int n = MR * RKP;
  if (i >= n) return;
  int r = i >> 6, k = i & 63;
  d[i] = (k < RK) ? f2b(proj[(size_t)r * PWD + k]) : f2b(0.f);
}

// dt_w [1536,48] -> bf16 [1536,64] zero-padded
__global__ void k_pad_dtw(const float* __restrict__ s, bf16* __restrict__ d) {
  int i = blockIdx.x * 256 + threadIdx.x;
  const int n = DI * RKP;
  if (i >= n) return;
  int r = i >> 6, k = i & 63;
  d[i] = (k < RK) ? f2b(s[(size_t)r * RK + k]) : f2b(0.f);
}

__global__ void k_softplus(const float* __restrict__ acc, const float* __restrict__ db,
                           float* __restrict__ dt) {
  int i = blockIdx.x * 256 + threadIdx.x;
  const int n = MR * DI;
  if (i >= n) return;
  float x = acc[i] + db[i % DI];
  dt[i] = (x > 20.f) ? x : log1pf(__expf(x));
}

// selective scan: one lane per channel, 16 states in registers, B/C via LDS.
__global__ __launch_bounds__(256)
void k_scan(const float* __restrict__ xc, const float* __restrict__ dt,
            const float* __restrict__ proj, const float* __restrict__ A_log,
            const float* __restrict__ Dp, const float* __restrict__ xz,
            float* __restrict__ y, int reverse) {
  __shared__ float sBC[2 * NS];
  const int b = blockIdx.x / (DI / 256);
  const int d = (blockIdx.x % (DI / 256)) * 256 + threadIdx.x;
  float A[NS], hs[NS];
#pragma unroll
  for (int nn = 0; nn < NS; nn++) {
    A[nn] = -__expf(A_log[(size_t)d * NS + nn]);
    hs[nn] = 0.f;
  }
  const float dpar = Dp[d];
  for (int l = 0; l < LSEQ; l++) {
    if (threadIdx.x < 2 * NS)
      sBC[threadIdx.x] = proj[(size_t)(b * LSEQ + l) * PWD + RK + threadIdx.x];
    __syncthreads();
    size_t idx = (size_t)(b * LSEQ + l) * DI + d;
    float dtv = dt[idx], xv = xc[idx], yv = 0.f;
#pragma unroll
    for (int nn = 0; nn < NS; nn++) {
      hs[nn] = __expf(dtv * A[nn]) * hs[nn] + dtv * sBC[nn] * xv;
      yv += hs[nn] * sBC[NS + nn];
    }
    yv += xv * dpar;
    int lo = reverse ? (LSEQ - 1 - l) : l;
    float z = xz[(size_t)(b * LSEQ + lo) * (2 * DI) + DI + d];
    float g = z / (1.f + __expf(-z));
    y[idx] = yv * g;
    __syncthreads();
  }
}

// ysum = bf16(y_f + reverse(y_b))
__global__ void k_combine(const float* __restrict__ yf, const float* __restrict__ yb,
                          bf16* __restrict__ ys) {
  int i = blockIdx.x * 256 + threadIdx.x;
  const int n = MR * DI;
  if (i >= n) return;
  int d = i % DI, l = (i / DI) % LSEQ, b = i / (DI * LSEQ);
  ys[i] = f2b(yf[i] + yb[(size_t)(b * LSEQ + (LSEQ - 1 - l)) * DI + d]);
}

// ============================================================
// Host side
// ============================================================
extern "C" void kernel_launch(void* const* d_in, const int* in_sizes, int n_in,
                              void* d_out, int out_size, void* d_ws, size_t ws_size,
                              hipStream_t stream) {
  (void)in_sizes; (void)n_in; (void)ws_size;
  const float* x_v      = (const float*)d_in[0];
  const float* x_a      = (const float*)d_in[1];
  const float* patch_vw = (const float*)d_in[2];
  const float* patch_vb = (const float*)d_in[3];
  const float* patch_aw = (const float*)d_in[4];
  const float* patch_ab = (const float*)d_in[5];
  const float* pos_v    = (const float*)d_in[6];
  const float* pos_a    = (const float*)d_in[7];
  const float* tpos_v   = (const float*)d_in[8];
  const float* tpos_a   = (const float*)d_in[9];
  const float* cls_v    = (const float*)d_in[10];
  const float* cls_a    = (const float*)d_in[11];
  const float* glob_v   = (const float*)d_in[12];
  const float* glob_a   = (const float*)d_in[13];
  const float* norm_w   = (const float*)d_in[14];
  const float* in_w     = (const float*)d_in[15];
  const float* conv_w   = (const float*)d_in[16];
  const float* conv_b   = (const float*)d_in[17];
  const float* conv_wb  = (const float*)d_in[18];
  const float* conv_bb  = (const float*)d_in[19];
  const float* xproj_w  = (const float*)d_in[20];
  const float* xproj_wb = (const float*)d_in[21];
  const float* dt_w     = (const float*)d_in[22];
  const float* dt_bias  = (const float*)d_in[23];
  const float* dt_wb    = (const float*)d_in[24];
  const float* dt_biasb = (const float*)d_in[25];
  const float* A_log    = (const float*)d_in[26];
  const float* A_logb   = (const float*)d_in[27];
  const float* D_par    = (const float*)d_in[28];
  const float* D_parb   = (const float*)d_in[29];
  const float* out_w    = (const float*)d_in[30];
  const int*   vis_idx  = (const int*)d_in[31];

  char* base = (char*)d_ws;
  size_t off = 0;
  auto alloc = [&](size_t bytes) -> void* {
    void* p = base + off;
    off = (off + bytes + 255) & ~(size_t)255;
    return p;
  };
  const int MV = BN * TT * NV;  // 1568
  const int MA = BN * 96;       // 192

  bf16*  Xv      = (bf16*)alloc((size_t)MV * 768 * 2);
  bf16*  Wv      = (bf16*)alloc((size_t)768 * 768 * 2);
  bf16*  Xa      = (bf16*)alloc((size_t)MA * 256 * 2);
  bf16*  Wa      = (bf16*)alloc((size_t)768 * 256 * 2);
  float* xv_emb  = (float*)alloc((size_t)MV * 768 * 4);
  float* xa_emb  = (float*)alloc((size_t)MA * 768 * 4);
  float* xfull   = (float*)alloc((size_t)BN * NTOT * 768 * 4);
  float* hbuf    = (float*)alloc((size_t)MR * 768 * 4);
  float* res     = (float*)alloc((size_t)MR * 768 * 4);
  bf16*  hn      = (bf16*)alloc((size_t)MR * 768 * 2);
  float* xz      = (float*)alloc((size_t)MR * 2 * DI * 4);
  float* xc      = (float*)alloc((size_t)MR * DI * 4);
  bf16*  xcb     = (bf16*)alloc((size_t)MR * DI * 2);
  float* proj    = (float*)alloc((size_t)MR * PWD * 4);
  bf16*  dtr_bf  = (bf16*)alloc((size_t)MR * RKP * 2);
  float* dtacc   = (float*)alloc((size_t)MR * DI * 4);
  float* dtbuf   = (float*)alloc((size_t)MR * DI * 4);
  float* y_f     = (float*)alloc((size_t)MR * DI * 4);
  float* y_b     = (float*)alloc((size_t)MR * DI * 4);
  bf16*  ysum    = (bf16*)alloc((size_t)MR * DI * 2);
  bf16*  inw_bf  = (bf16*)alloc((size_t)2 * DI * DM * 2);
  bf16*  xp_bf   = (bf16*)alloc((size_t)PWD * DI * 2);
  bf16*  dtw_bf  = (bf16*)alloc((size_t)DI * RKP * 2);
  bf16*  outw_bf = (bf16*)alloc((size_t)DM * DI * 2);

  const int TPB = 256;
  auto g1 = [&](int n) { return dim3((unsigned)((n + TPB - 1) / TPB)); };
  // generic: 1 tile per wave
  auto gemm = [&](const bf16* A, const bf16* W, const float* bias, float* C,
                  int M, int N, int K) {
    int tiles = (M / 16) * (N / 16);
    int blocks = (tiles + GEMM_WAVES - 1) / GEMM_WAVES;
    k_gemm_bf16<<<blocks, GEMM_WAVES * 32, 0, stream>>>(A, W, bias, C, M, N, K);
  };
  // 64x64 block per wave (M%64==0 && N%64==0)
  auto gemm44 = [&](const bf16* A, const bf16* W, const float* bias, float* C,
                    int M, int N, int K) {
    int blks = (M / 64) * (N / 64);
    int blocks = (blks + GEMM_WAVES - 1) / GEMM_WAVES;
    k_gemm_bf16_4x4<<<blocks, GEMM_WAVES * 32, 0, stream>>>(A, W, bias, C, M, N, K);
  };
  // TDM/LDS-staged 128x128 macro-tile per 4-wave block (M%128==0 && N%128==0)
  auto gemmT = [&](const bf16* A, const bf16* W, const float* bias, float* C,
                   int M, int N, int K) {
    int blocks = (M / 128) * (N / 128);
    k_gemm_bf16_tdm<<<blocks, 128, 0, stream>>>(A, W, bias, C, M, N, K);
  };

  // ---- patch embeds ----
  k_pack_video<<<g1(MV * 768), TPB, 0, stream>>>(x_v, Xv);
  k_pack_wv<<<g1(768 * 768), TPB, 0, stream>>>(patch_vw, Wv);
  k_pack_audio<<<g1(MA * 256), TPB, 0, stream>>>(x_a, Xa);
  k_f32_to_bf16<<<g1(768 * 256), TPB, 0, stream>>>(patch_aw, Wa, 768 * 256);
  gemm(Xv, Wv, patch_vb, xv_emb, MV, 768, 768);      // M=1568 -> generic
  gemm44(Xa, Wa, patch_ab, xa_emb, MA, 768, 256);    // M=192 -> 4x4

  // ---- tokens + visible gather ----
  k_assemble<<<g1(BN * NTOT * 768), TPB, 0, stream>>>(
      xv_emb, xa_emb, pos_v, pos_a, tpos_v, tpos_a, cls_v, cls_a, glob_v, glob_a, xfull);
  k_gather<<<g1(MR * 768), TPB, 0, stream>>>(xfull, vis_idx, hbuf, res);

  // ---- 16 bidirectional Mamba layers ----
  for (int L = 0; L < DEPTH; L++) {
    k_res_rmsnorm<<<MR, 256, 0, stream>>>(res, hbuf, norm_w + (size_t)L * DM, hn);
    k_f32_to_bf16<<<g1(2 * DI * DM), TPB, 0, stream>>>(
        in_w + (size_t)L * 2 * DI * DM, inw_bf, 2 * DI * DM);
    gemmT(hn, inw_bf, nullptr, xz, MR, 2 * DI, DM);

    for (int dir = 0; dir < 2; dir++) {
      const float* cw  = (dir ? conv_wb  : conv_w)  + (size_t)L * DI * KC;
      const float* cb  = (dir ? conv_bb  : conv_b)  + (size_t)L * DI;
      const float* xpw = (dir ? xproj_wb : xproj_w) + (size_t)L * PWD * DI;
      const float* dw  = (dir ? dt_wb    : dt_w)    + (size_t)L * DI * RK;
      const float* db  = (dir ? dt_biasb : dt_bias) + (size_t)L * DI;
      const float* al  = (dir ? A_logb   : A_log)   + (size_t)L * DI * NS;
      const float* dp  = (dir ? D_parb   : D_par)   + (size_t)L * DI;
      float* y = dir ? y_b : y_f;

      k_conv_silu<<<g1(MR * DI), TPB, 0, stream>>>(xz, cw, cb, xc, xcb, dir);
      k_f32_to_bf16<<<g1(PWD * DI), TPB, 0, stream>>>(xpw, xp_bf, PWD * DI);
      gemm(xcb, xp_bf, nullptr, proj, MR, PWD, DI);  // N=80 -> generic
      k_pad_dtr<<<g1(MR * RKP), TPB, 0, stream>>>(proj, dtr_bf);
      k_pad_dtw<<<g1(DI * RKP), TPB, 0, stream>>>(dw, dtw_bf);
      gemmT(dtr_bf, dtw_bf, nullptr, dtacc, MR, DI, RKP);
      k_softplus<<<g1(MR * DI), TPB, 0, stream>>>(dtacc, db, dtbuf);
      k_scan<<<BN * (DI / 256), 256, 0, stream>>>(xc, dtbuf, proj, al, dp, xz, y, dir);
    }

    k_combine<<<g1(MR * DI), TPB, 0, stream>>>(y_f, y_b, ysum);
    k_f32_to_bf16<<<g1(DM * DI), TPB, 0, stream>>>(
        out_w + (size_t)L * DM * DI, outw_bf, DM * DI);
    gemmT(ysum, outw_bf, nullptr, hbuf, MR, DM, DI);
  }

  // final residual stream is the reference's output
  k_copy_f32<<<g1(out_size), TPB, 0, stream>>>(res, (float*)d_out, out_size);
}
